// GptOssAttention_65738769433508
// MI455X (gfx1250) — matrix operand order repaired and checked by reading, not compiled, not run on hardware
//
#include <hip/hip_runtime.h>
#include <hip/hip_bf16.h>

// ---------------- types / helpers ----------------
typedef _Float16 h8   __attribute__((ext_vector_type(8)));
typedef _Float16 v16h __attribute__((ext_vector_type(16)));
typedef float    v8f  __attribute__((ext_vector_type(8)));

#define CAT16(lo, hi) __builtin_shufflevector((lo), (hi), 0,1,2,3,4,5,6,7,8,9,10,11,12,13,14,15)
#define WMMA(a, b, c) __builtin_amdgcn_wmma_f32_16x16x32_f16(false, (a), false, (b), (short)0, (c), false, false)

#define SLEN   2048
#define DMODEL 2048
#define NHEAD  32
#define HDIM   64
#define WINDOW 128

// ---------------- fp32 -> f16 convert ----------------
__global__ __launch_bounds__(256) void f32_to_f16_kernel(const float* __restrict__ in,
                                                         _Float16* __restrict__ out, int n) {
    int i = blockIdx.x * 256 + threadIdx.x;
    if (i < n) out[i] = (_Float16)in[i];
}

// ---------------- fp32 W[k][n] -> f16 Wt[n][k] ----------------
__global__ __launch_bounds__(256) void transpose_f16_kernel(const float* __restrict__ W,
                                                            _Float16* __restrict__ Wt) {
    __shared__ float tile[32][33];
    int bx = blockIdx.x * 32;      // k base
    int by = blockIdx.y * 32;      // n base
    int tx = threadIdx.x, ty = threadIdx.y;
#pragma unroll
    for (int i = ty; i < 32; i += 8)
        tile[i][tx] = W[(size_t)(bx + i) * DMODEL + by + tx];
    __syncthreads();
#pragma unroll
    for (int i = ty; i < 32; i += 8)
        Wt[(size_t)(by + i) * DMODEL + bx + tx] = (_Float16)tile[tx][i];
}

// ---------------- wave-level 32x64 WMMA GEMM core ----------------
// A = X (row-major f16, lda = DMODEL), B = Wt (row = out col, contiguous K).
__device__ __forceinline__ void wave_gemm_32x64(const _Float16* __restrict__ X,
                                                const _Float16* __restrict__ Wt,
                                                int m0, int n0, v8f (&acc)[2][4]) {
    int lane = threadIdx.x & 31;
    int ln = lane & 15, hi = lane >> 4;
#pragma unroll
    for (int mt = 0; mt < 2; ++mt)
#pragma unroll
        for (int t = 0; t < 4; ++t) acc[mt][t] = (v8f){0.f,0.f,0.f,0.f,0.f,0.f,0.f,0.f};

    const _Float16* arow0 = X + (size_t)(m0 + ln) * DMODEL;
    const _Float16* arow1 = X + (size_t)(m0 + 16 + ln) * DMODEL;

    for (int k0 = 0; k0 < DMODEL; k0 += 32) {
        v16h a[2];
        {
            const _Float16* p = arow0 + k0 + hi * 16;
            h8 lo = *(const h8*)p;
            h8 hb = *(const h8*)(p + 8);
            a[0] = CAT16(lo, hb);
            p = arow1 + k0 + hi * 16;
            lo = *(const h8*)p;
            hb = *(const h8*)(p + 8);
            a[1] = CAT16(lo, hb);
        }
#pragma unroll
        for (int t = 0; t < 4; ++t) {
            const _Float16* p = Wt + (size_t)(n0 + t * 16 + ln) * DMODEL + k0 + hi * 16;
            h8 lo = *(const h8*)p;
            h8 hb = *(const h8*)(p + 8);
            v16h b = CAT16(lo, hb);
            acc[0][t] = WMMA(a[0], b, acc[0][t]);
            acc[1][t] = WMMA(a[1], b, acc[1][t]);
        }
    }
}

// ---------------- QKV projection + RoPE ----------------
// mode 0: Q (rope + scale, out [h][s][64])
// mode 1: K (rope,          out [h][s][64])
// mode 2: V (no rope,       out [h][c][s]  -- transposed)
__global__ __launch_bounds__(128) void gemm_qkv_kernel(const _Float16* __restrict__ X,
                                                       const _Float16* __restrict__ Wt,
                                                       const float* __restrict__ cosb,
                                                       const float* __restrict__ sinb,
                                                       _Float16* __restrict__ outp, int mode) {
    int wave = threadIdx.x >> 5;
    int m0 = blockIdx.x * 128 + wave * 32;
    int h  = blockIdx.y;
    int n0 = h * HDIM;

    v8f acc[2][4];
    wave_gemm_32x64(X, Wt, m0, n0, acc);

    int lane = threadIdx.x & 31;
    int ln = lane & 15, hi = lane >> 4;
#pragma unroll
    for (int mt = 0; mt < 2; ++mt) {
#pragma unroll
        for (int r = 0; r < 8; ++r) {
            int s = m0 + mt * 16 + hi * 8 + r;
            if (mode == 2) {
#pragma unroll
                for (int t = 0; t < 4; ++t) {
                    int c = t * 16 + ln;
                    outp[(size_t)(h * HDIM + c) * SLEN + s] = (_Float16)acc[mt][t][r];
                }
            } else {
#pragma unroll
                for (int t = 0; t < 2; ++t) {
                    int j = t * 16 + ln;                 // 0..31 rotary index
                    float cs = cosb[(size_t)s * 32 + j];
                    float sn = sinb[(size_t)s * 32 + j];
                    float x1 = acc[mt][t][r];
                    float x2 = acc[mt][t + 2][r];
                    float o1 = x1 * cs - x2 * sn;
                    float o2 = x2 * cs + x1 * sn;
                    if (mode == 0) { o1 *= 0.125f; o2 *= 0.125f; }   // HD^-0.5 folded into Q
                    size_t base = ((size_t)h * SLEN + s) * HDIM;
                    outp[base + j]      = (_Float16)o1;
                    outp[base + 32 + j] = (_Float16)o2;
                }
            }
        }
    }
}

// ---------------- sliding-window attention with sink ----------------
__global__ __launch_bounds__(128) void attn_kernel(const _Float16* __restrict__ Qh,
                                                   const _Float16* __restrict__ Kh,
                                                   const _Float16* __restrict__ Vt,
                                                   const float* __restrict__ sinks,
                                                   _Float16* __restrict__ attnO) {
    __shared__ __align__(16) _Float16 probs[4][16][160];
    int lane = threadIdx.x & 31, wave = threadIdx.x >> 5;
    int ln = lane & 15, hi = lane >> 4;
    int h  = blockIdx.y;
    int q0 = blockIdx.x * 64 + wave * 16;

    // ---- load Q tile as A-matrix (16 x 64 -> two 16x32 chunks) ----
    const _Float16* qrow = Qh + ((size_t)h * SLEN + q0 + ln) * HDIM;
    v16h qa[2];
#pragma unroll
    for (int half = 0; half < 2; ++half) {
        h8 lo = *(const h8*)(qrow + half * 32 + hi * 16);
        h8 hb = *(const h8*)(qrow + half * 32 + hi * 16 + 8);
        qa[half] = CAT16(lo, hb);
    }

    // ---- phase 1: logits for 9 key tiles (keys q0-128 .. q0+15) ----
    v8f lg[9];
#pragma unroll
    for (int kt = 0; kt < 9; ++kt) {
        int kb  = q0 - WINDOW + kt * 16;
        int key = kb + ln;
        int keyc = key < 0 ? 0 : key;     // clamp; masked below
        const _Float16* krow = Kh + ((size_t)h * SLEN + keyc) * HDIM + hi * 16;
        h8 l0 = *(const h8*)(krow);
        h8 h0 = *(const h8*)(krow + 8);
        h8 l1 = *(const h8*)(krow + 32);
        h8 h1 = *(const h8*)(krow + 40);
        v16h b0 = CAT16(l0, h0);
        v16h b1 = CAT16(l1, h1);
        v8f c = (v8f){0.f,0.f,0.f,0.f,0.f,0.f,0.f,0.f};
        c = WMMA(qa[0], b0, c);
        c = WMMA(qa[1], b1, c);
#pragma unroll
        for (int r = 0; r < 8; ++r) {
            int q = q0 + hi * 8 + r;
            int diff = q - key;
            bool ok = (diff >= 0) && (diff < WINDOW) && (key >= 0);
            lg[kt][r] = ok ? c[r] : -1.0e9f;
        }
    }

    // ---- phase 2: softmax over 144 keys + sink column ----
    float sh = sinks[h];
#pragma unroll
    for (int r = 0; r < 8; ++r) {
        float mx = lg[0][r];
#pragma unroll
        for (int kt = 1; kt < 9; ++kt) mx = fmaxf(mx, lg[kt][r]);
#pragma unroll
        for (int off = 8; off >= 1; off >>= 1) mx = fmaxf(mx, __shfl_xor(mx, off, 32));
        mx = fmaxf(mx, sh);
        float sum = 0.f;
#pragma unroll
        for (int kt = 0; kt < 9; ++kt) {
            float e = __expf(lg[kt][r] - mx);
            lg[kt][r] = e;
            sum += e;
        }
#pragma unroll
        for (int off = 8; off >= 1; off >>= 1) sum += __shfl_xor(sum, off, 32);
        sum += __expf(sh - mx);              // sink in denominator only
        float inv = 1.0f / sum;
        int m = hi * 8 + r;
#pragma unroll
        for (int kt = 0; kt < 9; ++kt)
            probs[wave][m][kt * 16 + ln] = (_Float16)(lg[kt][r] * inv);
        probs[wave][m][144 + ln] = (_Float16)0.f;   // zero pad to 160 cols
    }
    // intra-wave LDS: DS ops are in-order per wave, no barrier needed.

    // ---- phase 3: O = P (16x160) @ V (160x64), 5 K-chunks of 32 ----
    v8f oacc[4];
#pragma unroll
    for (int ct = 0; ct < 4; ++ct) oacc[ct] = (v8f){0.f,0.f,0.f,0.f,0.f,0.f,0.f,0.f};
#pragma unroll
    for (int kk = 0; kk < 5; ++kk) {
        const _Float16* prow = &probs[wave][ln][0];
        h8 p0 = *(const h8*)(prow + kk * 32 + hi * 16);
        h8 p1 = *(const h8*)(prow + kk * 32 + hi * 16 + 8);
        v16h pa = CAT16(p0, p1);
        int kb0   = q0 - WINDOW + kk * 32;   // wave-uniform
        int kbase = kb0 + hi * 16;
        bool inb = (kb0 >= 0) && (kb0 + 31 < SLEN);   // uniform -> no EXEC divergence
#pragma unroll
        for (int ct = 0; ct < 4; ++ct) {
            const _Float16* vrow = Vt + (size_t)(h * HDIM + ct * 16 + ln) * SLEN;
            v16h vb;
            if (inb) {
                h8 v0 = *(const h8*)(vrow + kbase);
                h8 v1 = *(const h8*)(vrow + kbase + 8);
                vb = CAT16(v0, v1);
            } else {
#pragma unroll
                for (int e = 0; e < 16; ++e) {
                    int key = kbase + e;
                    vb[e] = (key >= 0 && key < SLEN) ? vrow[key] : (_Float16)0.f;
                }
            }
            oacc[ct] = WMMA(pa, vb, oacc[ct]);
        }
    }

    // ---- store O (f16 row-major [s][h*64+c]) for the output GEMM ----
#pragma unroll
    for (int ct = 0; ct < 4; ++ct)
#pragma unroll
        for (int r = 0; r < 8; ++r) {
            int s = q0 + hi * 8 + r;
            attnO[(size_t)s * DMODEL + h * HDIM + ct * 16 + ln] = (_Float16)oacc[ct][r];
        }
}

// ---------------- output projection: out = attn @ Wo (fp32 out) ----------------
__global__ __launch_bounds__(128) void gemm_out_kernel(const _Float16* __restrict__ A,
                                                       const _Float16* __restrict__ WoT,
                                                       float* __restrict__ out) {
    int wave = threadIdx.x >> 5;
    int m0 = blockIdx.x * 128 + wave * 32;
    int n0 = blockIdx.y * 64;

    v8f acc[2][4];
    wave_gemm_32x64(A, WoT, m0, n0, acc);

    int lane = threadIdx.x & 31;
    int ln = lane & 15, hi = lane >> 4;
#pragma unroll
    for (int mt = 0; mt < 2; ++mt)
#pragma unroll
        for (int t = 0; t < 4; ++t)
#pragma unroll
            for (int r = 0; r < 8; ++r) {
                int s = m0 + mt * 16 + hi * 8 + r;
                out[(size_t)s * DMODEL + n0 + t * 16 + ln] = acc[mt][t][r];
            }
}

// ---------------- launcher ----------------
extern "C" void kernel_launch(void* const* d_in, const int* in_sizes, int n_in,
                              void* d_out, int out_size, void* d_ws, size_t ws_size,
                              hipStream_t stream) {
    const float* hidden = (const float*)d_in[0];
    const float* cosb   = (const float*)d_in[1];
    const float* sinb   = (const float*)d_in[2];
    const float* Wq     = (const float*)d_in[3];
    const float* Wk     = (const float*)d_in[4];
    const float* Wv     = (const float*)d_in[5];
    const float* Wo     = (const float*)d_in[6];
    const float* sinks  = (const float*)d_in[7];
    float* out = (float*)d_out;

    char* ws = (char*)d_ws;
    const size_t MB = 1ull << 20;
    _Float16* Xh    = (_Float16*)(ws +  0 * MB);   // 8 MB  hidden f16
    _Float16* WqT   = (_Float16*)(ws +  8 * MB);   // 8 MB
    _Float16* WkT   = (_Float16*)(ws + 16 * MB);   // 8 MB
    _Float16* WvT   = (_Float16*)(ws + 24 * MB);   // 8 MB
    _Float16* WoT   = (_Float16*)(ws + 32 * MB);   // 8 MB
    _Float16* Qh    = (_Float16*)(ws + 40 * MB);   // 8 MB  [h][s][64]
    _Float16* Khd   = (_Float16*)(ws + 48 * MB);   // 8 MB  [h][s][64]
    _Float16* Vt    = (_Float16*)(ws + 56 * MB);   // 8 MB  [h][c][s]
    _Float16* attnH = (_Float16*)(ws + 64 * MB);   // 8 MB  [s][h*64+c]

    const int nElem = SLEN * DMODEL;
    f32_to_f16_kernel<<<nElem / 256, 256, 0, stream>>>(hidden, Xh, nElem);

    dim3 tb(32, 8), tg(DMODEL / 32, DMODEL / 32);
    transpose_f16_kernel<<<tg, tb, 0, stream>>>(Wq, WqT);
    transpose_f16_kernel<<<tg, tb, 0, stream>>>(Wk, WkT);
    transpose_f16_kernel<<<tg, tb, 0, stream>>>(Wv, WvT);
    transpose_f16_kernel<<<tg, tb, 0, stream>>>(Wo, WoT);

    dim3 gq(SLEN / 128, NHEAD);
    gemm_qkv_kernel<<<gq, 128, 0, stream>>>(Xh, WqT, cosb, sinb, Qh,  0);
    gemm_qkv_kernel<<<gq, 128, 0, stream>>>(Xh, WkT, cosb, sinb, Khd, 1);
    gemm_qkv_kernel<<<gq, 128, 0, stream>>>(Xh, WvT, cosb, sinb, Vt,  2);

    attn_kernel<<<dim3(SLEN / 64, NHEAD), 128, 0, stream>>>(Qh, Khd, Vt, sinks, attnH);

    gemm_out_kernel<<<dim3(SLEN / 128, DMODEL / 64), 128, 0, stream>>>(attnH, WoT, out);
}
